// GraphConvolutionalImputationGenerator_70463233458740
// MI455X (gfx1250) — matrix operand order repaired and checked by reading, not compiled, not run on hardware
//
#include <hip/hip_runtime.h>
#include <math.h>

// ---------------------------------------------------------------------------
// Problem constants (from the reference)
// ---------------------------------------------------------------------------
namespace {
constexpr int cB  = 4;
constexpr int cS  = 512;
constexpr int cG  = 48;
constexpr int cC  = 12;
constexpr int cD  = cG * cC;        // 576
constexpr int cN  = cB * cS * cG;   // 98304 nodes
constexpr int cE  = cN * 16;        // 1572864 edges
constexpr int cH  = 3;
constexpr int cFF = 2048;
constexpr int cHD = cD / cH;        // 192
constexpr int cRows = cB * cS;      // 2048 (= M for all big GEMMs)
constexpr float cScale = 0.07216878364870323f; // 1/sqrt(192)
}

typedef __attribute__((ext_vector_type(16))) __bf16 v16bf;
typedef __attribute__((ext_vector_type(8)))  float  v8f;
typedef __attribute__((ext_vector_type(8)))  unsigned short us8;

union Frag { v16bf v; us8 h[2]; };

// fp32 pair -> packed bf16 dword: round-to-nearest via +0x8000, then a single
// v_perm_b32 extracts both high halves ({b.hi16, a.hi16}).
__device__ __forceinline__ unsigned pack2(float a, float b) {
    unsigned ua = __float_as_uint(a) + 0x8000u;
    unsigned ub = __float_as_uint(b) + 0x8000u;
    return __builtin_amdgcn_perm(ub, ua, 0x07060302u);
}

// order-preserving float <-> int encoding for atomicMax on floats
__device__ __forceinline__ int   encF(float f) { int b = __float_as_int(f); return b ^ ((b >> 31) & 0x7fffffff); }
__device__ __forceinline__ float decF(int i)   { int b = i ^ ((i >> 31) & 0x7fffffff); return __int_as_float(b); }

// ---------------------------------------------------------------------------
// Generic batched bf16 WMMA GEMM:  C = A (MxK) x op(B) + bias, optional ReLU
//   A row-major, row stride lda.  transB=1: B is [N,K]; transB=0: B is [K,N].
//   Batch z: offset = (z/batDiv)*Out + (z%batDiv)*In  for A, B, C.
//   Requires: M%64==0, N%64==0, K%32==0; row strides multiple of 4 floats.
//   Block: 256 threads = 8 waves on a 64x64 tile; per wave: one A fragment x
//   two B fragments -> 2x v_wmma_f32_16x16x32_bf16 per K-step.
//   Double-buffered LDS (2x 4KB per operand) with one barrier per K-step:
//     issue next-tile global loads -> ds_load_b128 fragments + WMMA ->
//     pack/store staged regs into alternate buffer -> __syncthreads().
// ---------------------------------------------------------------------------
__global__ __launch_bounds__(256) void k_gemm_bf16(
    const float* __restrict__ A, long lda, long aOut, long aIn,
    const float* __restrict__ Bm, long ldb, long bOut, long bIn, int transB,
    float* __restrict__ Cm, long ldc, long cOut, long cIn,
    int M, int N, int K, int batDiv,
    const float* __restrict__ bias, int relu)
{
    __shared__ unsigned short la[2][64 * 32];   // [row][k]
    __shared__ unsigned short lb[2][64 * 32];   // [n][k]

    const int tid  = threadIdx.x;
    const int z    = blockIdx.z;
    const long offA = (long)(z / batDiv) * aOut + (long)(z % batDiv) * aIn;
    const long offB = (long)(z / batDiv) * bOut + (long)(z % batDiv) * bIn;
    const long offC = (long)(z / batDiv) * cOut + (long)(z % batDiv) * cIn;

    const int rowBase = blockIdx.y * 64;
    const int colBase = blockIdx.x * 64;

    const int wid  = tid >> 5;
    const int lane = tid & 31;
    const int wRow = (wid >> 1) << 4;   // 0,16,32,48
    const int wCol = (wid & 1) << 5;    // 0,32
    const int m16  = lane & 15;
    const int half = lane >> 4;

    // staging coordinates: each thread moves 8 elements of A and 8 of B
    const int ar = tid >> 2;            // 0..63  A row
    const int ak = (tid & 3) * 8;       // 0,8,16,24
    const int bn = tid >> 2;            // 0..63  B n
    const int bk = (tid & 3) * 8;

    const float* aBase = A  + offA + (long)(rowBase + ar) * lda + ak;
    const float* bBaseT = Bm + offB + (long)(colBase + bn) * ldb + bk; // transB=1
    const float* bBaseN = Bm + offB + (long)bk * ldb + (colBase + bn); // transB=0

    float av[8], bv[8];
    auto loadTiles = [&](int k0) {
        {
            const float4 t0 = *(const float4*)(aBase + k0);
            const float4 t1 = *(const float4*)(aBase + k0 + 4);
            av[0] = t0.x; av[1] = t0.y; av[2] = t0.z; av[3] = t0.w;
            av[4] = t1.x; av[5] = t1.y; av[6] = t1.z; av[7] = t1.w;
        }
        if (transB) {
            const float4 t0 = *(const float4*)(bBaseT + k0);
            const float4 t1 = *(const float4*)(bBaseT + k0 + 4);
            bv[0] = t0.x; bv[1] = t0.y; bv[2] = t0.z; bv[3] = t0.w;
            bv[4] = t1.x; bv[5] = t1.y; bv[6] = t1.z; bv[7] = t1.w;
        } else {
            const float* s = bBaseN + (long)k0 * ldb;
            #pragma unroll
            for (int j = 0; j < 8; ++j) bv[j] = s[(long)j * ldb];
        }
    };
    auto storeTiles = [&](int buf) {
        uint4 pa;
        pa.x = pack2(av[0], av[1]); pa.y = pack2(av[2], av[3]);
        pa.z = pack2(av[4], av[5]); pa.w = pack2(av[6], av[7]);
        *(uint4*)&la[buf][ar * 32 + ak] = pa;
        uint4 pb;
        pb.x = pack2(bv[0], bv[1]); pb.y = pack2(bv[2], bv[3]);
        pb.z = pack2(bv[4], bv[5]); pb.w = pack2(bv[6], bv[7]);
        *(uint4*)&lb[buf][bn * 32 + bk] = pb;
    };

    v8f acc0 = {0.f, 0.f, 0.f, 0.f, 0.f, 0.f, 0.f, 0.f};
    v8f acc1 = {0.f, 0.f, 0.f, 0.f, 0.f, 0.f, 0.f, 0.f};

    const int nk = K >> 5;
    loadTiles(0);
    storeTiles(0);
    __syncthreads();

    for (int i = 0; i < nk; ++i) {
        const int cur  = i & 1;
        const bool more = (i + 1) < nk;
        if (more) loadTiles((i + 1) << 5);     // next tile in flight under WMMA
        if (i + 2 < nk)                        // hint tile i+2 (global_prefetch_b8)
            __builtin_prefetch(aBase + ((i + 2) << 5), 0, 1);

        // fragments (contiguous k order, shared by A and B): A lane ks = 16h..16h+15, B ks = 16h..16h+15
        Frag fa, fb0, fb1;
        const unsigned short* pa = &la[cur][(wRow + m16) * 32 + 16 * half];
        fa.h[0] = *(const us8*)pa;
        fa.h[1] = *(const us8*)(pa + 8);
        const unsigned short* pb0 = &lb[cur][(wCol + m16) * 32 + 16 * half];
        fb0.h[0] = *(const us8*)pb0;
        fb0.h[1] = *(const us8*)(pb0 + 8);
        const unsigned short* pb1 = pb0 + 16 * 32;
        fb1.h[0] = *(const us8*)pb1;
        fb1.h[1] = *(const us8*)(pb1 + 8);

        acc0 = __builtin_amdgcn_wmma_f32_16x16x32_bf16(
                   false, fa.v, false, fb0.v, (short)0, acc0, false, false);
        acc1 = __builtin_amdgcn_wmma_f32_16x16x32_bf16(
                   false, fa.v, false, fb1.v, (short)0, acc1, false, false);

        if (more) storeTiles(cur ^ 1);
        __syncthreads();
    }

    const int n0 = colBase + wCol + m16;
    #pragma unroll
    for (int r = 0; r < 8; ++r) {
        int mrow = rowBase + wRow + r + 8 * half;
        float v0 = acc0[r];
        float v1 = acc1[r];
        if (bias) { v0 += bias[n0]; v1 += bias[n0 + 16]; }
        if (relu) { v0 = v0 > 0.f ? v0 : 0.f; v1 = v1 > 0.f ? v1 : 0.f; }
        Cm[offC + (long)mrow * ldc + n0]      = v0;
        Cm[offC + (long)mrow * ldc + n0 + 16] = v1;
    }
}

// ---------------------------------------------------------------------------
// Elementwise / GAT / softmax / layernorm / im2col kernels
// ---------------------------------------------------------------------------
__global__ __launch_bounds__(256) void k_blend(
    const float* __restrict__ x, const float* __restrict__ mask,
    const float* __restrict__ noise, float* __restrict__ out, int total)
{
    int i = blockIdx.x * 256 + threadIdx.x;
    if (i < total) out[i] = x[i] * mask[i] + noise[i] * (1.f - mask[i]);
}

// per-node: xp = h @ W^T ; es = xp.a_src ; ed = xp.a_dst ; zero accumulators
__global__ __launch_bounds__(256) void k_gat_prep(
    const float* __restrict__ h, const float* __restrict__ W,
    const float* __restrict__ a_s, const float* __restrict__ a_d,
    float* __restrict__ xp, float* __restrict__ es, float* __restrict__ ed,
    float* __restrict__ acc, float* __restrict__ denom, int* __restrict__ mmax)
{
    __shared__ float sW[cC * cC], sas[cC], sad[cC];
    int tid = threadIdx.x;
    if (tid < cC * cC) sW[tid] = W[tid];
    if (tid < cC) { sas[tid] = a_s[tid]; sad[tid] = a_d[tid]; }
    __syncthreads();
    int n = blockIdx.x * 256 + tid;
    if (n >= cN) return;
    float hv[cC];
    #pragma unroll
    for (int c = 0; c < cC; ++c) hv[c] = h[(long)n * cC + c];
    float s = 0.f, d = 0.f;
    #pragma unroll
    for (int co = 0; co < cC; ++co) {
        float a = 0.f;
        #pragma unroll
        for (int ci = 0; ci < cC; ++ci) a += sW[co * cC + ci] * hv[ci];
        xp[(long)n * cC + co] = a;
        acc[(long)n * cC + co] = 0.f;
        s += a * sas[co];
        d += a * sad[co];
    }
    es[n] = s; ed[n] = d; denom[n] = 0.f; mmax[n] = 0x80000000;
}

__global__ __launch_bounds__(256) void k_edge_max(
    const int* __restrict__ ei, const float* __restrict__ es,
    const float* __restrict__ ed, float* __restrict__ ebuf, int* __restrict__ mmax)
{
    int e = blockIdx.x * 256 + threadIdx.x;
    if (e >= cE + cN) return;
    int s = (e < cE) ? ei[e]      : (e - cE);
    int d = (e < cE) ? ei[cE + e] : (e - cE);
    float v = es[s] + ed[d];
    v = v > 0.f ? v : 0.2f * v;          // leaky relu 0.2
    ebuf[e] = v;
    atomicMax(mmax + d, encF(v));
}

__global__ __launch_bounds__(256) void k_edge_sum(
    const int* __restrict__ ei, float* __restrict__ ebuf,
    const int* __restrict__ mmax, float* __restrict__ denom)
{
    int e = blockIdx.x * 256 + threadIdx.x;
    if (e >= cE + cN) return;
    int d = (e < cE) ? ei[cE + e] : (e - cE);
    float w = __expf(ebuf[e] - decF(mmax[d]));
    ebuf[e] = w;
    atomicAdd(denom + d, w);
}

__global__ __launch_bounds__(256) void k_edge_acc(
    const int* __restrict__ ei, const float* __restrict__ ebuf,
    const float* __restrict__ denom, const float* __restrict__ xp,
    float* __restrict__ out)
{
    int e = blockIdx.x * 256 + threadIdx.x;
    if (e >= cE + cN) return;
    int s = (e < cE) ? ei[e]      : (e - cE);
    int d = (e < cE) ? ei[cE + e] : (e - cE);
    float alpha = ebuf[e] / denom[d];
    const float* xs = xp + (long)s * cC;
    float* o = out + (long)d * cC;
    #pragma unroll
    for (int c = 0; c < cC; ++c) atomicAdd(o + c, alpha * xs[c]);
}

__global__ __launch_bounds__(256) void k_bias_addC(
    float* __restrict__ t, const float* __restrict__ bias, int total)
{
    int i = blockIdx.x * 256 + threadIdx.x;
    if (i < total) t[i] += bias[i % cC];
}

__global__ __launch_bounds__(256) void k_softmax_row(float* __restrict__ sc, float scale)
{
    __shared__ float red[256];
    int tid = threadIdx.x;
    float* p = sc + (long)blockIdx.x * cS;
    float m = -3.0e38f;
    for (int j = tid; j < cS; j += 256) m = fmaxf(m, p[j] * scale);
    red[tid] = m; __syncthreads();
    for (int o = 128; o > 0; o >>= 1) { if (tid < o) red[tid] = fmaxf(red[tid], red[tid + o]); __syncthreads(); }
    m = red[0]; __syncthreads();
    float sum = 0.f;
    for (int j = tid; j < cS; j += 256) { float v = __expf(p[j] * scale - m); p[j] = v; sum += v; }
    red[tid] = sum; __syncthreads();
    for (int o = 128; o > 0; o >>= 1) { if (tid < o) red[tid] += red[tid + o]; __syncthreads(); }
    float inv = 1.f / red[0];
    for (int j = tid; j < cS; j += 256) p[j] *= inv;
}

// out = LayerNorm(x + r) * g + b   over rows of length D
__global__ __launch_bounds__(256) void k_add_ln(
    const float* __restrict__ x, const float* __restrict__ r,
    const float* __restrict__ g, const float* __restrict__ b,
    float* __restrict__ out)
{
    __shared__ float s1[256], s2[256];
    int tid = threadIdx.x;
    long row = blockIdx.x;
    const float* px = x + row * cD;
    const float* pr = r + row * cD;
    float ls = 0.f, lq = 0.f;
    for (int j = tid; j < cD; j += 256) { float v = px[j] + pr[j]; ls += v; lq += v * v; }
    s1[tid] = ls; s2[tid] = lq; __syncthreads();
    for (int o = 128; o > 0; o >>= 1) {
        if (tid < o) { s1[tid] += s1[tid + o]; s2[tid] += s2[tid + o]; }
        __syncthreads();
    }
    float mean = s1[0] / (float)cD;
    float var  = s2[0] / (float)cD - mean * mean;
    float inv  = rsqrtf(var + 1e-5f);
    for (int j = tid; j < cD; j += 256) {
        float v = px[j] + pr[j];
        out[row * cD + j] = (v - mean) * inv * g[j] + b[j];
    }
}

// im2col: in [Bt, Lin, Cin] -> col [Bt*Lout, Cin*3], k = ci*3 + tap
__global__ __launch_bounds__(256) void k_im2col(
    const float* __restrict__ in, float* __restrict__ col,
    int Lin, int Lout, int Cin, int stride, int pad, int total)
{
    int idx = blockIdx.x * 256 + threadIdx.x;
    if (idx >= total) return;
    int Kk = Cin * 3;
    int m = idx / Kk;
    int k = idx - m * Kk;
    int ci = k / 3, tap = k - ci * 3;
    int bb = m / Lout, l = m - bb * Lout;
    int li = l * stride + tap - pad;
    float v = 0.f;
    if (li >= 0 && li < Lin) v = in[((long)bb * Lin + li) * Cin + ci];
    col[idx] = v;
}

__global__ __launch_bounds__(256) void k_upsample2(
    const float* __restrict__ bt, float* __restrict__ ubt, int total)
{
    int idx = blockIdx.x * 256 + threadIdx.x;
    if (idx >= total) return;
    int ch = idx % (2 * cD);
    int row = idx / (2 * cD);              // b*512 + l
    int bb = row / cS, l = row - bb * cS;
    ubt[idx] = bt[((long)bb * (cS / 2) + (l >> 1)) * (2 * cD) + ch];
}

__global__ __launch_bounds__(256) void k_concat(
    const float* __restrict__ up, const float* __restrict__ e1,
    float* __restrict__ cat, int total)
{
    int idx = blockIdx.x * 256 + threadIdx.x;
    if (idx >= total) return;
    int ch = idx % (2 * cD);
    int row = idx / (2 * cD);
    cat[idx] = (ch < cD) ? up[(long)row * cD + ch] : e1[(long)row * cD + (ch - cD)];
}

__global__ __launch_bounds__(256) void k_finalize_tanh(
    const float* __restrict__ acc, const float* __restrict__ bias,
    float* __restrict__ out, int total)
{
    int i = blockIdx.x * 256 + threadIdx.x;
    if (i < total) out[i] = tanhf(acc[i] + bias[i % cC]);
}

// ---------------------------------------------------------------------------
// Host launcher
// ---------------------------------------------------------------------------
static void run_gemm(hipStream_t st,
                     const float* A, long lda, long aOut, long aIn,
                     const float* B, long ldb, long bOut, long bIn, int transB,
                     float* C, long ldc, long cOut, long cIn,
                     int M, int N, int K, int batches, int batDiv,
                     const float* bias, int relu)
{
    dim3 grid(N / 64, M / 64, batches), block(256);
    k_gemm_bf16<<<grid, block, 0, st>>>(A, lda, aOut, aIn, B, ldb, bOut, bIn, transB,
                                        C, ldc, cOut, cIn, M, N, K, batDiv, bias, relu);
}

extern "C" void kernel_launch(void* const* d_in, const int* in_sizes, int n_in,
                              void* d_out, int out_size, void* d_ws, size_t ws_size,
                              hipStream_t stream)
{
    (void)in_sizes; (void)n_in; (void)out_size; (void)ws_size;
    int a = 0;
    const int*   ei    = (const int*)  d_in[a++];
    const float* x     = (const float*)d_in[a++];
    const float* mask  = (const float*)d_in[a++];
    const float* noise = (const float*)d_in[a++];
    const float* Wenc  = (const float*)d_in[a++];
    const float* asE   = (const float*)d_in[a++];
    const float* adE   = (const float*)d_in[a++];
    const float* bEnc  = (const float*)d_in[a++];
    const float* Wq = (const float*)d_in[a++]; const float* bq = (const float*)d_in[a++];
    const float* Wk = (const float*)d_in[a++]; const float* bk = (const float*)d_in[a++];
    const float* Wv = (const float*)d_in[a++]; const float* bv = (const float*)d_in[a++];
    const float* Wo = (const float*)d_in[a++]; const float* bo = (const float*)d_in[a++];
    const float* ln1g = (const float*)d_in[a++]; const float* ln1b = (const float*)d_in[a++];
    const float* W1 = (const float*)d_in[a++]; const float* b1 = (const float*)d_in[a++];
    const float* W2 = (const float*)d_in[a++]; const float* b2 = (const float*)d_in[a++];
    const float* ln2g = (const float*)d_in[a++]; const float* ln2b = (const float*)d_in[a++];
    const float* e1w = (const float*)d_in[a++]; const float* e1b = (const float*)d_in[a++];
    const float* dww = (const float*)d_in[a++]; const float* dwb = (const float*)d_in[a++];
    const float* btw = (const float*)d_in[a++]; const float* btb = (const float*)d_in[a++];
    const float* upw = (const float*)d_in[a++]; const float* upb = (const float*)d_in[a++];
    const float* dcw = (const float*)d_in[a++]; const float* dcb = (const float*)d_in[a++];
    const float* Wdec = (const float*)d_in[a++];
    const float* asD  = (const float*)d_in[a++];
    const float* adD  = (const float*)d_in[a++];
    const float* bDec = (const float*)d_in[a++];

    // ---- workspace carve (bump allocator, 256B aligned) ----
    char* wp = (char*)d_ws;
    auto alloc = [&](size_t bytes) -> void* {
        void* p = (void*)wp;
        wp += (bytes + 255) & ~(size_t)255;
        return p;
    };
    const size_t NCf = (size_t)cN * cC * sizeof(float);        // node features
    const size_t RDf = (size_t)cRows * cD * sizeof(float);     // [2048,576]
    float* h0    = (float*)alloc(NCf);
    float* xp1   = (float*)alloc(NCf);
    float* es1   = (float*)alloc(cN * sizeof(float));
    float* ed1   = (float*)alloc(cN * sizeof(float));
    int*   mm1   = (int*)  alloc(cN * sizeof(int));
    float* den1  = (float*)alloc(cN * sizeof(float));
    float* ebuf  = (float*)alloc((size_t)(cE + cN) * sizeof(float));
    float* tbuf  = (float*)alloc(RDf);                          // GAT enc out == t
    float* qbuf  = (float*)alloc(RDf);
    float* kbuf  = (float*)alloc(RDf);
    float* vbuf  = (float*)alloc(RDf);
    float* scores= (float*)alloc((size_t)cB * cH * cS * cS * sizeof(float));
    float* obuf  = (float*)alloc(RDf);
    float* oproj = (float*)alloc(RDf);
    float* t1    = (float*)alloc(RDf);
    float* ffbuf = (float*)alloc((size_t)cRows * cFF * sizeof(float));
    float* ff2   = (float*)alloc(RDf);
    float* t2    = (float*)alloc(RDf);
    float* colbf = (float*)alloc((size_t)cRows * (2 * cD * 3) * sizeof(float)); // max im2col
    float* e1a   = (float*)alloc(RDf);
    float* d1a   = (float*)alloc((size_t)(cRows / 2) * 2 * cD * sizeof(float));
    float* bta   = (float*)alloc((size_t)(cRows / 2) * 2 * cD * sizeof(float));
    float* ubta  = (float*)alloc((size_t)cRows * 2 * cD * sizeof(float));
    float* upa   = (float*)alloc(RDf);
    float* cata  = (float*)alloc((size_t)cRows * 2 * cD * sizeof(float));
    float* ybuf  = (float*)alloc(RDf);
    float* xp2   = (float*)alloc(NCf);
    float* es2   = (float*)alloc(cN * sizeof(float));
    float* ed2   = (float*)alloc(cN * sizeof(float));
    int*   mm2   = (int*)  alloc(cN * sizeof(int));
    float* den2  = (float*)alloc(cN * sizeof(float));
    float* acc2  = (float*)alloc(NCf);

    float* out = (float*)d_out;

    const int NC  = cN * cC;
    const int EN  = cE + cN;
    dim3 blk(256);
    auto g1 = [](int total) { return dim3((total + 255) / 256); };

    // 1) masked blend
    k_blend<<<g1(NC), blk, 0, stream>>>(x, mask, noise, h0, NC);

    // 2) GAT encoder
    k_gat_prep<<<g1(cN), blk, 0, stream>>>(h0, Wenc, asE, adE, xp1, es1, ed1, tbuf, den1, mm1);
    k_edge_max<<<g1(EN), blk, 0, stream>>>(ei, es1, ed1, ebuf, mm1);
    k_edge_sum<<<g1(EN), blk, 0, stream>>>(ei, ebuf, mm1, den1);
    k_edge_acc<<<g1(EN), blk, 0, stream>>>(ei, ebuf, den1, xp1, tbuf);
    k_bias_addC<<<g1(NC), blk, 0, stream>>>(tbuf, bEnc, NC);

    // 3) Transformer layer  (tbuf viewed as [2048, 576])
    run_gemm(stream, tbuf, cD, 0, 0, Wq, cD, 0, 0, 1, qbuf, cD, 0, 0, cRows, cD, cD, 1, 1, bq, 0);
    run_gemm(stream, tbuf, cD, 0, 0, Wk, cD, 0, 0, 1, kbuf, cD, 0, 0, cRows, cD, cD, 1, 1, bk, 0);
    run_gemm(stream, tbuf, cD, 0, 0, Wv, cD, 0, 0, 1, vbuf, cD, 0, 0, cRows, cD, cD, 1, 1, bv, 0);
    // scores[b,h] = Q[b,:,h,:] @ K[b,:,h,:]^T   (batched over B*H)
    run_gemm(stream, qbuf, cD, (long)cS * cD, cHD,
                     kbuf, cD, (long)cS * cD, cHD, 1,
                     scores, cS, (long)cH * cS * cS, (long)cS * cS,
                     cS, cS, cHD, cB * cH, cH, nullptr, 0);
    k_softmax_row<<<dim3(cB * cH * cS), blk, 0, stream>>>(scores, cScale);
    // O[b,:,h,:] = att @ V[b,:,h,:]
    run_gemm(stream, scores, cS, (long)cH * cS * cS, (long)cS * cS,
                     vbuf, cD, (long)cS * cD, cHD, 0,
                     obuf, cD, (long)cS * cD, cHD,
                     cS, cHD, cS, cB * cH, cH, nullptr, 0);
    run_gemm(stream, obuf, cD, 0, 0, Wo, cD, 0, 0, 1, oproj, cD, 0, 0, cRows, cD, cD, 1, 1, bo, 0);
    k_add_ln<<<dim3(cRows), blk, 0, stream>>>(tbuf, oproj, ln1g, ln1b, t1);
    run_gemm(stream, t1, cD, 0, 0, W1, cD, 0, 0, 1, ffbuf, cFF, 0, 0, cRows, cFF, cD, 1, 1, b1, 1);
    run_gemm(stream, ffbuf, cFF, 0, 0, W2, cFF, 0, 0, 1, ff2, cD, 0, 0, cRows, cD, cFF, 1, 1, b2, 0);
    k_add_ln<<<dim3(cRows), blk, 0, stream>>>(t1, ff2, ln2g, ln2b, t2);

    // 4) UNet1d in [B, L, C] layout (im2col + WMMA GEMM per conv)
    int tot;
    tot = cRows * cD * 3;                 // e1 im2col: [2048, 1728]
    k_im2col<<<g1(tot), blk, 0, stream>>>(t2, colbf, cS, cS, cD, 1, 1, tot);
    run_gemm(stream, colbf, cD * 3, 0, 0, e1w, cD * 3, 0, 0, 1, e1a, cD, 0, 0,
             cRows, cD, cD * 3, 1, 1, e1b, 1);
    tot = (cRows / 2) * cD * 3;           // down im2col: [1024, 1728], stride 2, pad_left 0
    k_im2col<<<g1(tot), blk, 0, stream>>>(e1a, colbf, cS, cS / 2, cD, 2, 0, tot);
    run_gemm(stream, colbf, cD * 3, 0, 0, dww, cD * 3, 0, 0, 1, d1a, 2 * cD, 0, 0,
             cRows / 2, 2 * cD, cD * 3, 1, 1, dwb, 1);
    tot = (cRows / 2) * 2 * cD * 3;       // bott im2col: [1024, 3456]
    k_im2col<<<g1(tot), blk, 0, stream>>>(d1a, colbf, cS / 2, cS / 2, 2 * cD, 1, 1, tot);
    run_gemm(stream, colbf, 2 * cD * 3, 0, 0, btw, 2 * cD * 3, 0, 0, 1, bta, 2 * cD, 0, 0,
             cRows / 2, 2 * cD, 2 * cD * 3, 1, 1, btb, 1);
    tot = cRows * 2 * cD;                 // repeat x2 upsample
    k_upsample2<<<g1(tot), blk, 0, stream>>>(bta, ubta, tot);
    tot = cRows * 2 * cD * 3;             // up im2col: [2048, 3456]
    k_im2col<<<g1(tot), blk, 0, stream>>>(ubta, colbf, cS, cS, 2 * cD, 1, 1, tot);
    run_gemm(stream, colbf, 2 * cD * 3, 0, 0, upw, 2 * cD * 3, 0, 0, 1, upa, cD, 0, 0,
             cRows, cD, 2 * cD * 3, 1, 1, upb, 1);
    tot = cRows * 2 * cD;
    k_concat<<<g1(tot), blk, 0, stream>>>(upa, e1a, cata, tot);
    tot = cRows * 2 * cD * 3;             // dec im2col: [2048, 3456]
    k_im2col<<<g1(tot), blk, 0, stream>>>(cata, colbf, cS, cS, 2 * cD, 1, 1, tot);
    run_gemm(stream, colbf, 2 * cD * 3, 0, 0, dcw, 2 * cD * 3, 0, 0, 1, ybuf, cD, 0, 0,
             cRows, cD, 2 * cD * 3, 1, 1, dcb, 0);

    // 5) GAT decoder + tanh  (ybuf [B,S,D] == [N,C] contiguously)
    k_gat_prep<<<g1(cN), blk, 0, stream>>>(ybuf, Wdec, asD, adD, xp2, es2, ed2, acc2, den2, mm2);
    k_edge_max<<<g1(EN), blk, 0, stream>>>(ei, es2, ed2, ebuf, mm2);
    k_edge_sum<<<g1(EN), blk, 0, stream>>>(ei, ebuf, mm2, den2);
    k_edge_acc<<<g1(EN), blk, 0, stream>>>(ei, ebuf, den2, xp2, acc2);
    k_finalize_tanh<<<g1(NC), blk, 0, stream>>>(acc2, bDec, out, NC);
}
